// QwenForSentimentClassification_16234976379028
// MI455X (gfx1250) — matrix operand order repaired and checked by reading, not compiled, not run on hardware
//
#include <hip/hip_runtime.h>
#include <hip/hip_bf16.h>
#include <cmath>

typedef __attribute__((ext_vector_type(16))) __bf16 bf16x16;
typedef __attribute__((ext_vector_type(8)))  float  f32x8;
typedef unsigned short u16;
typedef __attribute__((__vector_size__(4 * sizeof(int)))) int v4i;
typedef __attribute__((address_space(1))) v4i* gv4i_p;
typedef __attribute__((address_space(3))) v4i* lv4i_p;

union Frag { uint4 u[2]; bf16x16 v; };

__device__ __forceinline__ unsigned f2bf(float f) {
  unsigned u = __builtin_bit_cast(unsigned, f);
  return (u + 0x7FFFu + ((u >> 16) & 1u)) >> 16;
}
__device__ __forceinline__ float bf2f(u16 b) {
  return __builtin_bit_cast(float, (unsigned)b << 16);
}

__device__ __forceinline__ void wait_async0() {
#if __has_builtin(__builtin_amdgcn_s_wait_asynccnt)
  __builtin_amdgcn_s_wait_asynccnt(0);
#elif defined(__gfx1250__)
  asm volatile("s_wait_asynccnt 0x0" ::: "memory");
#endif
}

// ---------------------------------------------------------------------------
// General batched bf16 GEMM: C[z] (+)= alpha * A[z] x B[z] (+ bias)
// A: bf16, strideM = lda, strideK = 1 (required). B: bf16, strides (ldbk, ldbn).
// WMMA f32_16x16x32_bf16, f32 accumulate. Output f32 (optionally +=) or bf16.
// Block: 256 thr (8 waves, 2x4), tile 64(M) x 128(N), K step 32.
// Requires M%64==0, N%128==0, K%32==0 (true for all shapes in this model).
// Batch z: zb=z/NHp, zh=z%NHp; B uses kv-head zh/rep (GQA).
// A tile staged to LDS via GLOBAL_LOAD_ASYNC_TO_LDS_B128 when available.
// ---------------------------------------------------------------------------
__global__ __launch_bounds__(256) void k_gemm(
    const u16* __restrict__ A, long lda, long aSB, long aSH,
    const u16* __restrict__ Bm, long ldbk, long ldbn, long bSB, long bSH, int rep,
    void* __restrict__ Cout, long ldc, long cSB, long cSH, int obf,
    const float* __restrict__ bias, float alpha, int acc, int K, int NHp)
{
  __shared__ __align__(16) u16 sA[64 * 40];   // 64 rows x 32 bf16, row pad -> 40
  __shared__ __align__(16) u16 sB[128 * 40];  // transposed [n][k], row pad -> 40

  const int z  = blockIdx.z;
  const int zb = z / NHp, zh = z % NHp;
  A  += (long)zb * aSB + (long)zh * aSH;
  Bm += (long)zb * bSB + (long)(zh / rep) * bSH;

  const int tid  = threadIdx.x;
  const int lane = tid & 31;
  const int wave = tid >> 5;
  const int wm = wave >> 2, wn = wave & 3;  // 2 x 4 waves
  const long m0 = (long)blockIdx.y * 64;
  const long n0 = (long)blockIdx.x * 128;

  f32x8 cacc[2][2] = {};

  const int ar  = tid >> 2;              // A tile row 0..63
  const int ac0 = (tid & 3) << 3;        // A tile col 0,8,16,24
  const int bk0 = (tid >> 4) << 1;       // B tile rows k, k+1 (0..30)
  const int bn0 = (tid & 15) << 3;       // B tile cols 0..120 step 8

  for (int k0 = 0; k0 < K; k0 += 32) {
    // ---- stage A tile 64x32 bf16 (straight copy), 8 elems (16B) / thread ----
    {
      const u16* ap = A + (m0 + ar) * lda + k0 + ac0;
#if __has_builtin(__builtin_amdgcn_global_load_async_to_lds_b128)
      __builtin_amdgcn_global_load_async_to_lds_b128(
          (gv4i_p)(u16*)ap, (lv4i_p)(&sA[ar * 40 + ac0]), 0, 0);
#else
      *(uint4*)(&sA[ar * 40 + ac0]) = *(const uint4*)ap;
#endif
    }
    // ---- stage B tile 32x128 bf16 transposed into sB[n][k] ----
    {
      const u16* bp0 = Bm + (long)(k0 + bk0) * ldbk + (n0 + bn0) * ldbn;
      const u16* bp1 = bp0 + ldbk;
      union { uint4 q; u16 s[8]; } r0, r1;
      if (ldbn == 1) {
        r0.q = *(const uint4*)bp0;
        r1.q = *(const uint4*)bp1;
      } else {
#pragma unroll
        for (int i = 0; i < 8; i++) {
          r0.s[i] = bp0[(long)i * ldbn];
          r1.s[i] = bp1[(long)i * ldbn];
        }
      }
#pragma unroll
      for (int i = 0; i < 8; i++) {
        unsigned pk = (unsigned)r0.s[i] | ((unsigned)r1.s[i] << 16);
        *(unsigned*)(&sB[(bn0 + i) * 40 + bk0]) = pk;
      }
    }
    // prefetch next B K-tile while computing this one
    if (k0 + 32 < K) {
      __builtin_prefetch(Bm + (long)(k0 + 32 + bk0) * ldbk + (n0 + bn0) * ldbn, 0, 1);
    }
    wait_async0();
    __syncthreads();

    // ---- load fragments + 4 WMMAs ----
    Frag bfr[2];
#pragma unroll
    for (int nf = 0; nf < 2; nf++) {
      // B frag 32x16: lane n=L%16, base=(L/16)*8; VGPR0-3: K=base..base+7,
      // VGPR4-7: K=16+base..16+base+7
      int col = wn * 32 + nf * 16 + (lane & 15);
      int kk  = (lane >> 4) << 3;  // 0 or 8
      const u16* pb = &sB[col * 40 + kk];
      bfr[nf].u[0] = *(const uint4*)pb;
      bfr[nf].u[1] = *(const uint4*)(pb + 16);
    }
#pragma unroll
    for (int mf = 0; mf < 2; mf++) {
      // A frag 16x32: lane m=L%16, base=(L/16)*8; VGPR0-3: K=base..base+7,
      // VGPR4-7: K=16+base..16+base+7
      int row = wm * 32 + mf * 16 + (lane & 15);
      int kk  = (lane >> 4) << 3;  // 0 or 8
      const u16* pa = &sA[row * 40 + kk];
      Frag af;
      af.u[0] = *(const uint4*)pa;
      af.u[1] = *(const uint4*)(pa + 16);
#pragma unroll
      for (int nf = 0; nf < 2; nf++) {
        cacc[mf][nf] = __builtin_amdgcn_wmma_f32_16x16x32_bf16(
            false, af.v, false, bfr[nf].v, (short)0, cacc[mf][nf], false, false);
      }
    }
    __syncthreads();
  }

  // ---- epilogue: C/D layout lane n=L%16, VGPR j -> m=(L/16)*8+j ----
  float* Cf  = (float*)Cout + (long)zb * cSB + (long)zh * cSH;
  u16*   Ch  = (u16*)Cout + (long)zb * cSB + (long)zh * cSH;
#pragma unroll
  for (int mf = 0; mf < 2; mf++) {
#pragma unroll
    for (int nf = 0; nf < 2; nf++) {
      long nn    = n0 + wn * 32 + nf * 16 + (lane & 15);
      long mbase = m0 + wm * 32 + mf * 16 + ((lane >> 4) << 3);
      float badd = bias ? bias[nn] : 0.0f;
#pragma unroll
      for (int j = 0; j < 8; j++) {
        long mm = mbase + j;
        float v = alpha * cacc[mf][nf][j] + badd;
        if (obf) {
          Ch[mm * ldc + nn] = (u16)f2bf(v);
        } else {
          float* cp = Cf + mm * ldc + nn;
          if (acc) v += *cp;
          *cp = v;
        }
      }
    }
  }
}

// ---------------------------------------------------------------------------
// Helper kernels
// ---------------------------------------------------------------------------
__global__ void k_cvt(const float* __restrict__ in, u16* __restrict__ out, long n) {
  long t = ((long)blockIdx.x * blockDim.x + threadIdx.x) * 4;
  if (t >= n) return;
  float4 f = *(const float4*)(in + t);
  uint2 pk;
  pk.x = f2bf(f.x) | (f2bf(f.y) << 16);
  pk.y = f2bf(f.z) | (f2bf(f.w) << 16);
  *(uint2*)(out + t) = pk;
}

__global__ void k_rope_tables(float* __restrict__ cosT, float* __restrict__ sinT,
                              int S, int DHd) {
  int t = blockIdx.x * blockDim.x + threadIdx.x;
  int half = DHd / 2;
  if (t >= S * half) return;
  int s = t / half, i = t % half;
  float inv = expf(-logf(1000000.0f) * (2.0f * i) / (float)DHd);
  float f = (float)s * inv;
  float c = cosf(f), sn = sinf(f);
  cosT[s * DHd + i] = c;  cosT[s * DHd + half + i] = c;
  sinT[s * DHd + i] = sn; sinT[s * DHd + half + i] = sn;
}

__global__ __launch_bounds__(256) void k_embed(const float* __restrict__ emb,
                                               const int* __restrict__ ids,
                                               float* __restrict__ h, int Hd) {
  long t = blockIdx.x;
  long id = ids[t];
  const float4* src = (const float4*)(emb + id * (long)Hd);
  float4* dst = (float4*)(h + t * (long)Hd);
  for (int i = threadIdx.x; i < Hd / 4; i += 256) dst[i] = src[i];
}

// rmsnorm f32 -> bf16
__global__ __launch_bounds__(256) void k_rmsnorm(const float* __restrict__ x,
                                                 const float* __restrict__ w,
                                                 u16* __restrict__ out, int Hd) {
  __shared__ float red[256];
  long row = blockIdx.x;
  const float* xr = x + row * Hd;
  float s = 0.f;
  for (int i = threadIdx.x; i < Hd; i += 256) { float v = xr[i]; s += v * v; }
  red[threadIdx.x] = s; __syncthreads();
  for (int o = 128; o > 0; o >>= 1) {
    if (threadIdx.x < o) red[threadIdx.x] += red[threadIdx.x + o];
    __syncthreads();
  }
  float rs = rsqrtf(red[0] / (float)Hd + 1e-6f);
  for (int i = threadIdx.x; i < Hd; i += 256)
    out[row * Hd + i] = (u16)f2bf(xr[i] * rs * w[i]);
}

// t[m,r] = act(bf16)[m,:] . Am[r,:]
__global__ void k_lora_xa(const u16* __restrict__ act, const float* __restrict__ Am,
                          float* __restrict__ tl, int K, int Rr, long total) {
  long t = (long)blockIdx.x * blockDim.x + threadIdx.x;
  if (t >= total) return;
  int r = (int)(t % Rr);
  long m = t / Rr;
  const u16* xr = act + m * (long)K;
  const float* ar = Am + (long)r * K;
  float s = 0.f;
  for (int k = 0; k < K; k += 4)
    s += bf2f(xr[k]) * ar[k] + bf2f(xr[k + 1]) * ar[k + 1] +
         bf2f(xr[k + 2]) * ar[k + 2] + bf2f(xr[k + 3]) * ar[k + 3];
  tl[t] = s;
}

__global__ void k_lora_add(float* __restrict__ C, const float* __restrict__ tl,
                           const float* __restrict__ Bm, int N, float scale, long total) {
  long t = (long)blockIdx.x * blockDim.x + threadIdx.x;
  if (t >= total) return;
  long m = t / N;
  int n = (int)(t % N);
  const float* tr = tl + m * 16;
  const float* br = Bm + (long)n * 16;
  float s = 0.f;
#pragma unroll
  for (int r = 0; r < 16; r++) s += tr[r] * br[r];
  C[t] += scale * s;
}

// read f32, rotate, write bf16 copy
__global__ void k_rope_apply(const float* __restrict__ x, u16* __restrict__ xh,
                             const float* __restrict__ cosT, const float* __restrict__ sinT,
                             int nh, int S, int DHd, long total) {
  long t = (long)blockIdx.x * blockDim.x + threadIdx.x;
  if (t >= total) return;
  int half = DHd / 2;
  int d = (int)(t % half);
  long rem = t / half;
  int hh = (int)(rem % nh);
  long tok = rem / nh;
  int s = (int)(tok % S);
  long base = (tok * nh + hh) * (long)DHd;
  float x1 = x[base + d], x2 = x[base + half + d];
  float c = cosT[s * DHd + d], sn = sinT[s * DHd + d];
  xh[base + d] = (u16)f2bf(x1 * c - x2 * sn);
  xh[base + half + d] = (u16)f2bf(x2 * c + x1 * sn);
}

// masked causal softmax over bf16 rows, in place
__global__ __launch_bounds__(256) void k_softmax(u16* __restrict__ sc,
                                                 const int* __restrict__ mask,
                                                 int S, int NHp, long nrows) {
  int lane = threadIdx.x & 31;
  long wrow = (long)blockIdx.x * (blockDim.x >> 5) + (threadIdx.x >> 5);
  if (wrow >= nrows) return;
  int q = (int)(wrow % S);
  long bh = wrow / S;
  int b = (int)(bh / NHp);
  u16* row = sc + wrow * S;
  const int* mrow = mask + (long)b * S;
  float mx = -1e30f;
  for (int j = lane; j < S; j += 32) {
    bool ok = (j <= q) && (mrow[j] != 0);
    mx = fmaxf(mx, ok ? bf2f(row[j]) : -1e30f);
  }
  for (int o = 16; o > 0; o >>= 1) mx = fmaxf(mx, __shfl_xor(mx, o, 32));
  float sum = 0.f;
  float ev[16];
  int c = 0;
  for (int j = lane; j < S; j += 32, c++) {
    bool ok = (j <= q) && (mrow[j] != 0);
    float e = ok ? __expf(bf2f(row[j]) - mx) : 0.0f;
    sum += e;
    ev[c] = e;
  }
  for (int o = 16; o > 0; o >>= 1) sum += __shfl_xor(sum, o, 32);
  float inv = 1.0f / sum;
  c = 0;
  for (int j = lane; j < S; j += 32, c++) row[j] = (u16)f2bf(ev[c] * inv);
}

// g = bf16(silu(g) * u) in place, bf16 in/out
__global__ void k_swiglu(u16* __restrict__ g, const u16* __restrict__ u, long total) {
  long t = (long)blockIdx.x * blockDim.x + threadIdx.x;
  if (t >= total) return;
  float x = bf2f(g[t]);
  float uu = bf2f(u[t]);
  g[t] = (u16)f2bf((x / (1.0f + __expf(-x))) * uu);
}

__global__ __launch_bounds__(256) void k_pool_norm(const float* __restrict__ h,
                                                   const float* __restrict__ lnf,
                                                   const int* __restrict__ mask,
                                                   float* __restrict__ pooled,
                                                   int S, int Hd) {
  __shared__ float red[256];
  __shared__ int lastIdx;
  int b = blockIdx.x;
  int cnt = 0;
  for (int i = threadIdx.x; i < S; i += 256) cnt += mask[(long)b * S + i];
  red[threadIdx.x] = (float)cnt; __syncthreads();
  for (int o = 128; o > 0; o >>= 1) {
    if (threadIdx.x < o) red[threadIdx.x] += red[threadIdx.x + o];
    __syncthreads();
  }
  if (threadIdx.x == 0) lastIdx = (int)red[0] - 1;
  __syncthreads();
  const float* xr = h + ((long)b * S + lastIdx) * Hd;
  float s = 0.f;
  for (int i = threadIdx.x; i < Hd; i += 256) { float v = xr[i]; s += v * v; }
  __syncthreads();
  red[threadIdx.x] = s; __syncthreads();
  for (int o = 128; o > 0; o >>= 1) {
    if (threadIdx.x < o) red[threadIdx.x] += red[threadIdx.x + o];
    __syncthreads();
  }
  float rs = rsqrtf(red[0] / (float)Hd + 1e-6f);
  for (int i = threadIdx.x; i < Hd; i += 256) pooled[(long)b * Hd + i] = xr[i] * rs * lnf[i];
}

__global__ void k_fc1(const float* __restrict__ pooled, const float* __restrict__ w,
                      const float* __restrict__ bias, float* __restrict__ out,
                      int Hd, long total) {
  long t = (long)blockIdx.x * blockDim.x + threadIdx.x;
  if (t >= total) return;
  int j = (int)(t % Hd);
  long b = t / Hd;
  const float* pr = pooled + b * Hd;
  float s = bias[j];
  for (int k = 0; k < Hd; k++) s += pr[k] * w[(long)k * Hd + j];
  out[t] = fmaxf(s, 0.0f);
}

__global__ void k_fc2(const float* __restrict__ fc1, const float* __restrict__ w,
                      const float* __restrict__ bias, float* __restrict__ out,
                      int Hd, int NC, int total) {
  int t = blockIdx.x * blockDim.x + threadIdx.x;
  if (t >= total) return;
  int c = t % NC;
  int b = t / NC;
  const float* fr = fc1 + (long)b * Hd;
  float s = bias[c];
  for (int k = 0; k < Hd; k++) s += fr[k] * w[k * NC + c];
  out[t] = s;
}

// ---------------------------------------------------------------------------
extern "C" void kernel_launch(void* const* d_in, const int* in_sizes, int n_in,
                              void* d_out, int out_size, void* d_ws, size_t ws_size,
                              hipStream_t stream) {
  (void)in_sizes; (void)n_in; (void)out_size; (void)ws_size;
  const int Bb = 4, Ss = 512, Hh = 2048, NHq = 16, NKVh = 4, DHd = 128;
  const int Ff = 5632, Rr = 16, NCc = 2, Ll = 4;
  const int QD = NHq * DHd;   // 2048
  const int KD = NKVh * DHd;  // 512
  const int M = Bb * Ss;      // 2048
  const float LSC = 2.0f;     // alpha/r = 32/16

  const int*   ids   = (const int*)d_in[0];
  const int*   mask  = (const int*)d_in[1];
  const float* embed = (const float*)d_in[2];
  const float* ln1   = (const float*)d_in[3];
  const float* wq    = (const float*)d_in[4];
  const float* bq    = (const float*)d_in[5];
  const float* wk    = (const float*)d_in[6];
  const float* bk    = (const float*)d_in[7];
  const float* wv    = (const float*)d_in[8];
  const float* bv    = (const float*)d_in[9];
  const float* wo    = (const float*)d_in[10];
  const float* laq   = (const float*)d_in[11];
  const float* lbq   = (const float*)d_in[12];
  const float* lak   = (const float*)d_in[13];
  const float* lbk   = (const float*)d_in[14];
  const float* lav   = (const float*)d_in[15];
  const float* lbv   = (const float*)d_in[16];
  const float* lao   = (const float*)d_in[17];
  const float* lbo   = (const float*)d_in[18];
  const float* ln2   = (const float*)d_in[19];
  const float* wg    = (const float*)d_in[20];
  const float* wu    = (const float*)d_in[21];
  const float* wd    = (const float*)d_in[22];
  const float* lnf   = (const float*)d_in[23];
  const float* cw1   = (const float*)d_in[24];
  const float* cb1   = (const float*)d_in[25];
  const float* cw2   = (const float*)d_in[26];
  const float* cb2   = (const float*)d_in[27];

  char* wsp = (char*)d_ws;
  size_t off = 0;
  auto alloc = [&](size_t bytes) {
    void* p = (void*)(wsp + off);
    off += ((bytes + 255) / 256) * 256;
    return p;
  };
  float* cosT = (float*)alloc((size_t)Ss * DHd * 4);
  float* sinT = (float*)alloc((size_t)Ss * DHd * 4);
  float* h    = (float*)alloc((size_t)M * Hh * 4);
  u16* xn16   = (u16*)alloc((size_t)M * Hh * 2);
  float* qf   = (float*)alloc((size_t)M * QD * 4);
  float* kf   = (float*)alloc((size_t)M * KD * 4);
  float* vf   = (float*)alloc((size_t)M * KD * 4);
  u16* qh     = (u16*)alloc((size_t)M * QD * 2);
  u16* kh     = (u16*)alloc((size_t)M * KD * 2);
  u16* vh     = (u16*)alloc((size_t)M * KD * 2);
  u16* sch    = (u16*)alloc((size_t)Bb * NHq * Ss * Ss * 2);
  u16* ctxh   = (u16*)alloc((size_t)M * QD * 2);
  u16* gh     = (u16*)alloc((size_t)M * Ff * 2);
  u16* uh     = (u16*)alloc((size_t)M * Ff * 2);
  float* tl   = (float*)alloc((size_t)M * Rr * 4);
  float* pooled = (float*)alloc((size_t)Bb * Hh * 4);
  float* fc1b   = (float*)alloc((size_t)Bb * Hh * 4);
  u16* wq16 = (u16*)alloc((size_t)Hh * QD * 2);
  u16* wk16 = (u16*)alloc((size_t)Hh * KD * 2);
  u16* wv16 = (u16*)alloc((size_t)Hh * KD * 2);
  u16* wo16 = (u16*)alloc((size_t)QD * Hh * 2);
  u16* wg16 = (u16*)alloc((size_t)Hh * Ff * 2);
  u16* wu16 = (u16*)alloc((size_t)Hh * Ff * 2);
  u16* wd16 = (u16*)alloc((size_t)Ff * Hh * 2);

  auto cvt = [&](const float* in, u16* out, long n) {
    k_cvt<<<(unsigned)((n / 4 + 255) / 256), 256, 0, stream>>>(in, out, n);
  };
  auto gemm = [&](const u16* A, long lda, long aSB, long aSH,
                  const u16* Bm, long ldbk, long ldbn, long bSB, long bSH, int rep,
                  void* C, long ldc, long cSB, long cSH, int obf,
                  const float* bias, float alpha, int acc,
                  int Mm, int Nn, int Kk, int NHp, int nz) {
    dim3 g(Nn / 128, Mm / 64, nz);
    k_gemm<<<g, 256, 0, stream>>>(A, lda, aSB, aSH, Bm, ldbk, ldbn, bSB, bSH, rep,
                                  C, ldc, cSB, cSH, obf, bias, alpha, acc, Kk, NHp);
  };

  const float alphaS = 1.0f / sqrtf((float)DHd);

  k_rope_tables<<<(Ss * DHd / 2 + 255) / 256, 256, 0, stream>>>(cosT, sinT, Ss, DHd);
  k_embed<<<M, 256, 0, stream>>>(embed, ids, h, Hh);

  for (int l = 0; l < Ll; l++) {
    const float* wq_l = wq + (size_t)l * Hh * QD;
    const float* bq_l = bq + (size_t)l * QD;
    const float* wk_l = wk + (size_t)l * Hh * KD;
    const float* bk_l = bk + (size_t)l * KD;
    const float* wv_l = wv + (size_t)l * Hh * KD;
    const float* bv_l = bv + (size_t)l * KD;
    const float* wo_l = wo + (size_t)l * QD * Hh;
    const float* laq_l = laq + (size_t)l * Rr * Hh;
    const float* lbq_l = lbq + (size_t)l * QD * Rr;
    const float* lak_l = lak + (size_t)l * Rr * Hh;
    const float* lbk_l = lbk + (size_t)l * KD * Rr;
    const float* lav_l = lav + (size_t)l * Rr * Hh;
    const float* lbv_l = lbv + (size_t)l * KD * Rr;
    const float* lao_l = lao + (size_t)l * Rr * QD;
    const float* lbo_l = lbo + (size_t)l * Hh * Rr;
    const float* wg_l = wg + (size_t)l * Hh * Ff;
    const float* wu_l = wu + (size_t)l * Hh * Ff;
    const float* wd_l = wd + (size_t)l * Ff * Hh;

    // convert this layer's dense weights to bf16 (reused scratch)
    cvt(wq_l, wq16, (long)Hh * QD);
    cvt(wk_l, wk16, (long)Hh * KD);
    cvt(wv_l, wv16, (long)Hh * KD);
    cvt(wo_l, wo16, (long)QD * Hh);
    cvt(wg_l, wg16, (long)Hh * Ff);
    cvt(wu_l, wu16, (long)Hh * Ff);
    cvt(wd_l, wd16, (long)Ff * Hh);

    // attn input norm -> bf16
    k_rmsnorm<<<M, 256, 0, stream>>>(h, ln1 + (size_t)l * Hh, xn16, Hh);

    // Q/K/V projections (+bias) and LoRA (f32 outputs)
    gemm(xn16, Hh, 0, 0, wq16, QD, 1, 0, 0, 1, qf, QD, 0, 0, 0, bq_l, 1.0f, 0, M, QD, Hh, 1, 1);
    k_lora_xa<<<(M * Rr) / 256, 256, 0, stream>>>(xn16, laq_l, tl, Hh, Rr, (long)M * Rr);
    k_lora_add<<<((long)M * QD) / 256, 256, 0, stream>>>(qf, tl, lbq_l, QD, LSC, (long)M * QD);

    gemm(xn16, Hh, 0, 0, wk16, KD, 1, 0, 0, 1, kf, KD, 0, 0, 0, bk_l, 1.0f, 0, M, KD, Hh, 1, 1);
    k_lora_xa<<<(M * Rr) / 256, 256, 0, stream>>>(xn16, lak_l, tl, Hh, Rr, (long)M * Rr);
    k_lora_add<<<((long)M * KD) / 256, 256, 0, stream>>>(kf, tl, lbk_l, KD, LSC, (long)M * KD);

    gemm(xn16, Hh, 0, 0, wv16, KD, 1, 0, 0, 1, vf, KD, 0, 0, 0, bv_l, 1.0f, 0, M, KD, Hh, 1, 1);
    k_lora_xa<<<(M * Rr) / 256, 256, 0, stream>>>(xn16, lav_l, tl, Hh, Rr, (long)M * Rr);
    k_lora_add<<<((long)M * KD) / 256, 256, 0, stream>>>(vf, tl, lbv_l, KD, LSC, (long)M * KD);

    // RoPE q,k (f32 -> bf16); v -> bf16
    long tq = (long)M * NHq * (DHd / 2);
    k_rope_apply<<<(unsigned)((tq + 255) / 256), 256, 0, stream>>>(qf, qh, cosT, sinT, NHq, Ss, DHd, tq);
    long tk = (long)M * NKVh * (DHd / 2);
    k_rope_apply<<<(unsigned)((tk + 255) / 256), 256, 0, stream>>>(kf, kh, cosT, sinT, NKVh, Ss, DHd, tk);
    cvt(vf, vh, (long)M * KD);

    // scores = (q @ k^T)/sqrt(DH) -> bf16, batched over B*NH with GQA mapping
    gemm(qh, QD, (long)Ss * QD, DHd,
         kh, 1, KD, (long)Ss * KD, DHd, NHq / NKVh,
         sch, Ss, (long)NHq * Ss * Ss, (long)Ss * Ss, 1,
         nullptr, alphaS, 0, Ss, Ss, DHd, NHq, Bb * NHq);

    long nrows = (long)Bb * NHq * Ss;
    k_softmax<<<(unsigned)(nrows / 8), 256, 0, stream>>>(sch, mask, Ss, NHq, nrows);

    // ctx = attn @ v -> bf16
    gemm(sch, Ss, (long)NHq * Ss * Ss, (long)Ss * Ss,
         vh, KD, 1, (long)Ss * KD, DHd, NHq / NKVh,
         ctxh, QD, (long)Ss * QD, DHd, 1,
         nullptr, 1.0f, 0, Ss, DHd, Ss, NHq, Bb * NHq);

    // h += ctx @ wo (+ LoRA-o)
    gemm(ctxh, QD, 0, 0, wo16, Hh, 1, 0, 0, 1, h, Hh, 0, 0, 0, nullptr, 1.0f, 1, M, Hh, QD, 1, 1);
    k_lora_xa<<<(M * Rr) / 256, 256, 0, stream>>>(ctxh, lao_l, tl, QD, Rr, (long)M * Rr);
    k_lora_add<<<((long)M * Hh) / 256, 256, 0, stream>>>(h, tl, lbo_l, Hh, LSC, (long)M * Hh);

    // MLP: h += (silu(x@wg) * (x@wu)) @ wd
    k_rmsnorm<<<M, 256, 0, stream>>>(h, ln2 + (size_t)l * Hh, xn16, Hh);
    gemm(xn16, Hh, 0, 0, wg16, Ff, 1, 0, 0, 1, gh, Ff, 0, 0, 1, nullptr, 1.0f, 0, M, Ff, Hh, 1, 1);
    gemm(xn16, Hh, 0, 0, wu16, Ff, 1, 0, 0, 1, uh, Ff, 0, 0, 1, nullptr, 1.0f, 0, M, Ff, Hh, 1, 1);
    k_swiglu<<<(unsigned)(((long)M * Ff) / 256), 256, 0, stream>>>(gh, uh, (long)M * Ff);
    gemm(gh, Ff, 0, 0, wd16, Hh, 1, 0, 0, 1, h, Hh, 0, 0, 0, nullptr, 1.0f, 1, M, Hh, Ff, 1, 1);
  }

  // final norm at last valid token + classifier
  k_pool_norm<<<Bb, 256, 0, stream>>>(h, lnf, mask, pooled, Ss, Hh);
  k_fc1<<<(Bb * Hh) / 256, 256, 0, stream>>>(pooled, cw1, cb1, fc1b, Hh, (long)Bb * Hh);
  k_fc2<<<1, 64, 0, stream>>>(fc1b, cw2, cb2, (float*)d_out, Hh, NCc, Bb * NCc);
}
